// AdaptiveTrajectoryDecoder_48034914238991
// MI455X (gfx1250) — matrix operand order repaired and checked by reading, not compiled, not run on hardware
//
#include <hip/hip_runtime.h>
#include <hip/hip_bf16.h>

// ---------------- problem constants ----------------
static constexpr int BTOT = 65536;   // batch
static constexpr int HID  = 256;     // hidden
static constexpr int EMB  = 64;      // embed
static constexpr int DIM  = 2;       // position dim
static constexpr int STEPS= 12;      // P
static constexpr int KG   = EMB + HID;   // 320, gate GEMM K
static constexpr int ROWS = 32;      // batch rows per workgroup (2 WMMA M-tiles)
static constexpr int NWAVES = 16;
static constexpr int NTHREADS = NWAVES * 32;
static constexpr int XPAD = 328;     // x_lds row stride (bf16), 656B = 16B aligned
static constexpr int HHPAD = 128;    // head-hidden row stride (bf16)
static constexpr int NKC  = KG / 32;   // 10 gate K-chunks
static constexpr int NKH  = HID / 32;  // 8 head K-chunks

// ---------------- WMMA types ----------------
typedef __bf16 v16bf __attribute__((ext_vector_type(16)));
typedef float  v8f   __attribute__((ext_vector_type(8)));
typedef unsigned int u32x4 __attribute__((ext_vector_type(4)));

union Frag {
  v16bf v;
  u32x4 u[2];
};

// ---------------- helpers ----------------
__device__ __forceinline__ unsigned short f2bf_rne(float f) {
  unsigned int u = __float_as_uint(f);
  unsigned int r = u + 0x7FFFu + ((u >> 16) & 1u);
  return (unsigned short)(r >> 16);
}
__device__ __forceinline__ float bfu_lo(unsigned int u) {
  return __uint_as_float(u << 16);
}
__device__ __forceinline__ float bfu_hi(unsigned int u) {
  return __uint_as_float(u & 0xFFFF0000u);
}
__device__ __forceinline__ float sigmoid_f(float x) {
  return 1.0f / (1.0f + __expf(-x));
}
__device__ __forceinline__ float tanh_f(float x) {
  x = fminf(15.0f, fmaxf(-15.0f, x));
  float e = __expf(2.0f * x);
  return (e - 1.0f) / (e + 1.0f);
}

// ---------------- weight packing kernel ----------------
// Wg   [1024][320] bf16 : packed row p = w*64 + gate*16 + t  <->  orig row gate*256 + (w*16 + t)
//                         cols 0..63 = W_ih, cols 64..319 = W_hh
// biasg[1024] f32       : b_ih + b_hh, same packed order
// Whead[128][256] bf16  : rows 0..63 = W_sp1, rows 64..127 = W_un1
__global__ void prep_kernel(const float* __restrict__ Wih, const float* __restrict__ Whh,
                            const float* __restrict__ bih, const float* __restrict__ bhh,
                            const float* __restrict__ Wsp1, const float* __restrict__ Wun1,
                            unsigned short* __restrict__ Wg, float* __restrict__ biasg,
                            unsigned short* __restrict__ Whead) {
  int idx = blockIdx.x * blockDim.x + threadIdx.x;
  if (idx < 1024 * KG) {
    int p = idx / KG, k = idx - p * KG;
    int gate = (p >> 4) & 3;
    int w = p >> 6;
    int t = p & 15;
    int r = gate * HID + w * 16 + t;
    float v = (k < EMB) ? Wih[(size_t)r * EMB + k]
                        : Whh[(size_t)r * HID + (k - EMB)];
    Wg[idx] = f2bf_rne(v);
  }
  if (idx < 1024) {
    int gate = (idx >> 4) & 3, w = idx >> 6, t = idx & 15;
    int r = gate * HID + w * 16 + t;
    biasg[idx] = bih[r] + bhh[r];
  }
  if (idx < 128 * HID) {
    int q = idx / HID, k = idx - q * HID;
    float v = (q < 64) ? Wsp1[q * HID + k] : Wun1[(q - 64) * HID + k];
    Whead[idx] = f2bf_rne(v);
  }
}

// ---------------- main recurrent kernel ----------------
__global__ __launch_bounds__(NTHREADS)
void traj_kernel(const float* __restrict__ h0, const float* __restrict__ c0,
                 const float* __restrict__ pos0,
                 const float* __restrict__ W_embed, const float* __restrict__ b_embed,
                 const float* __restrict__ W_pos,   const float* __restrict__ b_pos,
                 const float* __restrict__ W_sp2,   const float* __restrict__ b_sp2,
                 const float* __restrict__ W_un2,   const float* __restrict__ b_un2,
                 const float* __restrict__ b_sp1,   const float* __restrict__ b_un1,
                 const unsigned short* __restrict__ Wg,
                 const float* __restrict__ biasg,
                 const unsigned short* __restrict__ Whead,
                 float* __restrict__ out_pred,      // [B][P][2]
                 float* __restrict__ out_sp,        // [B][P][1]
                 float* __restrict__ out_un) {      // [B][P][2]
  __shared__ unsigned short x_lds[ROWS][XPAD];      // [emb(64) | h(256)] bf16, 21KB
  __shared__ unsigned short hh_lds[ROWS][HHPAD];    // relu'd head hidden bf16, 8KB
  __shared__ float pos_lds[ROWS][DIM];
  __shared__ float wpos_lds[DIM][HID];
  __shared__ float wsp2_lds[64];
  __shared__ float wun2_lds[DIM][64];
  __shared__ float wemb_lds[EMB][DIM];
  __shared__ float bemb_lds[EMB];
  __shared__ float sc_lds[8];                       // [0..1]=b_pos, [2]=b_sp2, [3..4]=b_un2

  const int tid  = threadIdx.x;
  const int wave = tid >> 5;
  const int lane = tid & 31;
  const int lr   = lane & 15;
  const int lh   = lane >> 4;
  const int rowBase = blockIdx.x * ROWS;

  // ---- init: h -> LDS (bf16), pos -> LDS, small weights -> LDS ----
  for (int i = tid; i < ROWS * HID; i += NTHREADS) {
    int r = i >> 8, k = i & (HID - 1);
    x_lds[r][EMB + k] = f2bf_rne(h0[(size_t)(rowBase + r) * HID + k]);
  }
  for (int i = tid; i < DIM * HID; i += NTHREADS) {
    int d = i >> 8, k = i & (HID - 1);
    wpos_lds[d][k] = W_pos[i];
  }
  if (tid < ROWS * DIM) {
    int r = tid >> 1, d = tid & 1;
    pos_lds[r][d] = pos0[(size_t)(rowBase + r) * DIM + d];
  }
  if (tid < 64)        wsp2_lds[tid] = W_sp2[tid];
  if (tid < 128)       wun2_lds[tid >> 6][tid & 63] = W_un2[tid];
  if (tid < EMB * DIM) wemb_lds[tid >> 1][tid & 1] = W_embed[tid];
  if (tid < EMB)       bemb_lds[tid] = b_embed[tid];
  if (tid < 2) { sc_lds[tid] = b_pos[tid]; sc_lds[3 + tid] = b_un2[tid]; }
  if (tid == 2)  sc_lds[2] = b_sp2[0];

  // c -> registers in accumulator layout; wave owns hidden units [16w,16w+16)
  float creg[2][8];
  {
    int j = wave * 16 + lr;
    #pragma unroll
    for (int mt = 0; mt < 2; ++mt)
      #pragma unroll
      for (int v = 0; v < 8; ++v) {
        int r = mt * 16 + v + 8 * lh;
        creg[mt][v] = c0[(size_t)(rowBase + r) * HID + j];
      }
  }
  float gbias[4];
  #pragma unroll
  for (int g = 0; g < 4; ++g) gbias[g] = biasg[wave * 64 + g * 16 + lr];
  float hbias;
  {
    int hc = (wave & 7) * 16 + lr;
    hbias = (hc < 64) ? b_sp1[hc] : b_un1[hc - 64];
  }
  const int hmt = wave >> 3;
  const int hnt = wave & 7;

  const unsigned short* wgbase = Wg + (size_t)(wave * 64 + lr) * KG + lh * 8;
  const unsigned short* whbase = Whead + (size_t)(hnt * 16 + lr) * HID + lh * 8;

  __syncthreads();

  const v8f vzero = {0.f, 0.f, 0.f, 0.f, 0.f, 0.f, 0.f, 0.f};

  // double-buffered gate weight fragments (static buffers, no dynamic indexing)
  Frag b0[4], b1[4];
  auto loadB = [&](Frag (&dst)[4], int kc) {
    const int kb = kc * 32;
    #pragma unroll
    for (int g = 0; g < 4; ++g) {
      const unsigned short* bb = wgbase + (size_t)g * 16 * KG + kb;
      dst[g].u[0] = *(const u32x4*)(bb);
      dst[g].u[1] = *(const u32x4*)(bb + 16);
    }
  };
  Frag hb0, hb1;
  auto loadHB = [&](Frag& dst, int kc) {
    const unsigned short* bb = whbase + kc * 32;
    dst.u[0] = *(const u32x4*)(bb);
    dst.u[1] = *(const u32x4*)(bb + 16);
  };
  auto loadA2 = [&](Frag& a0, Frag& a1, int kc) {
    const int kb = kc * 32;
    const unsigned short* s0 = &x_lds[lr][kb + lh * 8];
    a0.u[0] = *(const u32x4*)(s0);
    a0.u[1] = *(const u32x4*)(s0 + 16);
    const unsigned short* s1 = &x_lds[16 + lr][kb + lh * 8];
    a1.u[0] = *(const u32x4*)(s1);
    a1.u[1] = *(const u32x4*)(s1 + 16);
  };

  // prime the weight pipelines once; the wrapped tail prefetch re-arms them each step
  loadB(b0, 0);
  loadB(b1, 1);
  loadHB(hb0, 0);
  loadHB(hb1, 1);

  #pragma unroll 1
  for (int p = 0; p < STEPS; ++p) {
    // ---- phase A: emb = relu(pos @ W_embed^T + b_embed) -> x_lds[:,0:64] ----
    for (int i = tid; i < ROWS * EMB; i += NTHREADS) {
      int r = i >> 6, e = i & 63;
      float v = pos_lds[r][0] * wemb_lds[e][0] +
                pos_lds[r][1] * wemb_lds[e][1] + bemb_lds[e];
      x_lds[r][e] = f2bf_rne(v > 0.f ? v : 0.f);
    }
    __syncthreads();

    // ---- phase B: gates[32,1024] = x[32,320] @ Wg^T (pipelined WMMA) ----
    v8f acc[2][4];
    #pragma unroll
    for (int mt = 0; mt < 2; ++mt)
      #pragma unroll
      for (int g = 0; g < 4; ++g) acc[mt][g] = vzero;

    #pragma unroll 1
    for (int kc = 0; kc < NKC; kc += 2) {
      Frag a0, a1;
      // even chunk: consume b0, prefetch (kc+2) mod NKC into b0
      loadA2(a0, a1, kc);
      #pragma unroll
      for (int g = 0; g < 4; ++g) {
        acc[0][g] = __builtin_amdgcn_wmma_f32_16x16x32_bf16(
            false, a0.v, false, b0[g].v, (short)0, acc[0][g], false, false);
        acc[1][g] = __builtin_amdgcn_wmma_f32_16x16x32_bf16(
            false, a1.v, false, b0[g].v, (short)0, acc[1][g], false, false);
      }
      {
        int nk = kc + 2; if (nk >= NKC) nk -= NKC;   // wraps to next step's chunk 0
        loadB(b0, nk);
      }
      // odd chunk: consume b1, prefetch (kc+3) mod NKC into b1
      loadA2(a0, a1, kc + 1);
      #pragma unroll
      for (int g = 0; g < 4; ++g) {
        acc[0][g] = __builtin_amdgcn_wmma_f32_16x16x32_bf16(
            false, a0.v, false, b1[g].v, (short)0, acc[0][g], false, false);
        acc[1][g] = __builtin_amdgcn_wmma_f32_16x16x32_bf16(
            false, a1.v, false, b1[g].v, (short)0, acc[1][g], false, false);
      }
      {
        int nk = kc + 3; if (nk >= NKC) nk -= NKC;   // wraps to next step's chunk 1
        loadB(b1, nk);
      }
    }
    __syncthreads();   // all x_lds reads done before h is overwritten

    // ---- phase C: lane-local LSTM cell (c in registers), write h bf16 ----
    #pragma unroll
    for (int mt = 0; mt < 2; ++mt) {
      #pragma unroll
      for (int v = 0; v < 8; ++v) {
        float iv = acc[mt][0][v] + gbias[0];
        float fv = acc[mt][1][v] + gbias[1];
        float gv = acc[mt][2][v] + gbias[2];
        float ov = acc[mt][3][v] + gbias[3];
        float cn = sigmoid_f(fv) * creg[mt][v] + sigmoid_f(iv) * tanh_f(gv);
        creg[mt][v] = cn;
        float hv = sigmoid_f(ov) * tanh_f(cn);
        int r = mt * 16 + v + 8 * lh;
        x_lds[r][EMB + wave * 16 + lr] = f2bf_rne(hv);
      }
    }
    __syncthreads();

    // ---- phase D: head hidden [32,128] = h[32,256] @ [W_sp1;W_un1]^T ----
    {
      v8f hacc = vzero;
      #pragma unroll 1
      for (int kc = 0; kc < NKH; kc += 2) {
        Frag a;
        const unsigned short* as = &x_lds[hmt * 16 + lr][EMB + kc * 32 + lh * 8];
        a.u[0] = *(const u32x4*)(as);
        a.u[1] = *(const u32x4*)(as + 16);
        hacc = __builtin_amdgcn_wmma_f32_16x16x32_bf16(
            false, a.v, false, hb0.v, (short)0, hacc, false, false);
        { int nk = kc + 2; if (nk >= NKH) nk -= NKH; loadHB(hb0, nk); }
        const unsigned short* as1 = &x_lds[hmt * 16 + lr][EMB + (kc + 1) * 32 + lh * 8];
        a.u[0] = *(const u32x4*)(as1);
        a.u[1] = *(const u32x4*)(as1 + 16);
        hacc = __builtin_amdgcn_wmma_f32_16x16x32_bf16(
            false, a.v, false, hb1.v, (short)0, hacc, false, false);
        { int nk = kc + 3; if (nk >= NKH) nk -= NKH; loadHB(hb1, nk); }
      }
      #pragma unroll
      for (int v = 0; v < 8; ++v) {
        float hv = hacc[v] + hbias;
        hv = hv > 0.f ? hv : 0.f;
        int r = hmt * 16 + v + 8 * lh;
        hh_lds[r][hnt * 16 + lr] = f2bf_rne(hv);
      }
    }
    __syncthreads();

    // ---- phase E: pred (all 512 threads, 8-lane split) + sp/un (vectorized) ----
    {
      int rd  = tid >> 3;
      int sub = tid & 7;
      int r = rd >> 1, d = rd & 1;
      float s = 0.f;
      const u32x4* hx = (const u32x4*)&x_lds[r][EMB + sub * 32];
      const float* wp = &wpos_lds[d][sub * 32];
      #pragma unroll
      for (int q = 0; q < 4; ++q) {
        u32x4 hv = hx[q];
        #pragma unroll
        for (int e = 0; e < 4; ++e) {
          unsigned int u = hv[e];
          s += bfu_lo(u) * wp[q * 8 + e * 2] + bfu_hi(u) * wp[q * 8 + e * 2 + 1];
        }
      }
      s += __shfl_xor(s, 1);
      s += __shfl_xor(s, 2);
      s += __shfl_xor(s, 4);
      if (sub == 0) {
        float pr = pos_lds[r][d] + s + sc_lds[d];
        out_pred[((size_t)(rowBase + r) * STEPS + p) * DIM + d] = pr;
        pos_lds[r][d] = pr;
      }
    }
    if (tid < 32) {
      int r = tid;
      float s = sc_lds[2];
      const u32x4* hh = (const u32x4*)&hh_lds[r][0];
      #pragma unroll
      for (int q = 0; q < 8; ++q) {
        u32x4 hv = hh[q];
        #pragma unroll
        for (int e = 0; e < 4; ++e) {
          unsigned int u = hv[e];
          s += bfu_lo(u) * wsp2_lds[q * 8 + e * 2] + bfu_hi(u) * wsp2_lds[q * 8 + e * 2 + 1];
        }
      }
      float sp = (s > 15.f) ? s : __logf(1.f + __expf(s));
      out_sp[(size_t)(rowBase + r) * STEPS + p] = sp;
    } else if (tid < 160) {
      int u2 = tid - 32;
      int r = u2 >> 1, d = u2 & 1;
      float s = sc_lds[3 + d];
      const u32x4* hh = (const u32x4*)&hh_lds[r][64];
      #pragma unroll
      for (int q = 0; q < 8; ++q) {
        u32x4 hv = hh[q];
        #pragma unroll
        for (int e = 0; e < 4; ++e) {
          unsigned int u = hv[e];
          s += bfu_lo(u) * wun2_lds[d][q * 8 + e * 2] + bfu_hi(u) * wun2_lds[d][q * 8 + e * 2 + 1];
        }
      }
      out_un[((size_t)(rowBase + r) * STEPS + p) * DIM + d] = __expf(s);
    }
    __syncthreads();
  }
}

// ---------------- launcher ----------------
extern "C" void kernel_launch(void* const* d_in, const int* in_sizes, int n_in,
                              void* d_out, int out_size, void* d_ws, size_t ws_size,
                              hipStream_t stream) {
  (void)in_sizes; (void)n_in; (void)out_size; (void)ws_size;
  const float* h0      = (const float*)d_in[0];
  const float* c0      = (const float*)d_in[1];
  const float* pos0    = (const float*)d_in[2];
  const float* W_embed = (const float*)d_in[3];
  const float* b_embed = (const float*)d_in[4];
  const float* W_ih    = (const float*)d_in[5];
  const float* W_hh    = (const float*)d_in[6];
  const float* b_ih    = (const float*)d_in[7];
  const float* b_hh    = (const float*)d_in[8];
  const float* W_pos   = (const float*)d_in[9];
  const float* b_pos   = (const float*)d_in[10];
  const float* W_sp1   = (const float*)d_in[11];
  const float* b_sp1   = (const float*)d_in[12];
  const float* W_sp2   = (const float*)d_in[13];
  const float* b_sp2   = (const float*)d_in[14];
  const float* W_un1   = (const float*)d_in[15];
  const float* b_un1   = (const float*)d_in[16];
  const float* W_un2   = (const float*)d_in[17];
  const float* b_un2   = (const float*)d_in[18];

  unsigned char* ws = (unsigned char*)d_ws;
  unsigned short* Wg    = (unsigned short*)ws;                                  // 655360 B
  float*          biasg = (float*)(ws + (size_t)1024 * KG * 2);                 // 4096 B
  unsigned short* Whead = (unsigned short*)(ws + (size_t)1024 * KG * 2 + 4096); // 65536 B

  int prepN = 1024 * KG;
  prep_kernel<<<(prepN + 255) / 256, 256, 0, stream>>>(
      W_ih, W_hh, b_ih, b_hh, W_sp1, W_un1, Wg, biasg, Whead);

  float* out_pred = (float*)d_out;
  float* out_sp   = out_pred + (size_t)BTOT * STEPS * DIM;
  float* out_un   = out_sp   + (size_t)BTOT * STEPS;

  traj_kernel<<<BTOT / ROWS, NTHREADS, 0, stream>>>(
      h0, c0, pos0, W_embed, b_embed, W_pos, b_pos, W_sp2, b_sp2,
      W_un2, b_un2, b_sp1, b_un1, Wg, biasg, Whead,
      out_pred, out_sp, out_un);
}
